// WeatherLSTM_57638461112752
// MI455X (gfx1250) — matrix operand order repaired and checked by reading, not compiled, not run on hardware
//
#include <hip/hip_runtime.h>

typedef __attribute__((ext_vector_type(16))) _Float16 v16h;
typedef __attribute__((ext_vector_type(8)))  _Float16 v8h;
typedef __attribute__((ext_vector_type(8)))  float    v8f;

#define T_STEPS 168
#define IN_F    4
#define HID     64
#define G4      256   // 4*HID gate columns
#define HSTR    72    // f16 row stride for h buffers (144B = 36 banks -> conflict-free)
#define GSTR    276   // f32 row stride for gate buffer (276%64=20, 8 rows -> +32 banks)

union V16H { v16h v; _Float16 e[16]; };
union V8F  { v8f  v; float    e[8];  };

__device__ __forceinline__ v8f wmma16(v16h a, v16h b, v8f c) {
  // D = A(16x32 f16) * B(32x16 f16) + C(16x16 f32)
  return __builtin_amdgcn_wmma_f32_16x16x32_f16(false, a, false, b, (short)0, c,
                                                false, false);
}

// Build one K=32 A-chunk from an LDS h buffer (16-bit A layout):
// lane m = lane&15; half j -> K = kbase + hi*16 + j (contiguous).
__device__ __forceinline__ v16h ld_a(const _Float16* hb, int m, int hi8, int kbase) {
  V16H u;
  const _Float16* p = hb + m * HSTR + kbase + 2 * hi8;
  *(v8h*)&u.e[0] = *(const v8h*)(p);
  *(v8h*)&u.e[8] = *(const v8h*)(p + 8);
  return u.v;
}

__device__ __forceinline__ float fsigmoid(float x) {
  return 1.0f / (1.0f + __expf(-x));
}
__device__ __forceinline__ float ftanh(float x) {
  // 1 - 2/(e^{2x}+1): saturates cleanly to +/-1 without inf/inf NaNs
  return 1.0f - 2.0f / (__expf(2.0f * x) + 1.0f);
}

__global__ __launch_bounds__(256)
void lstm2_fused_kernel(const float* __restrict__ x,
                        const float* __restrict__ Wih0, const float* __restrict__ Whh0,
                        const float* __restrict__ bih0, const float* __restrict__ bhh0,
                        const float* __restrict__ Wih1, const float* __restrict__ Whh1,
                        const float* __restrict__ bih1, const float* __restrict__ bhh1,
                        const float* __restrict__ W1,   const float* __restrict__ b1,
                        const float* __restrict__ W2,   const float* __restrict__ b2,
                        float* __restrict__ out)
{
  __shared__ _Float16 sh0[16 * HSTR];          // layer-0 hidden state (f16 = WMMA A source)
  __shared__ _Float16 sh1[16 * HSTR];          // layer-1 hidden state
  __shared__ _Float16 sxa[T_STEPS * 16 * IN_F];// whole x tile for this block, f16
  __shared__ float    sG [16 * GSTR];          // gate pre-activations (f32), reused for z

  const int tid  = threadIdx.x;
  const int lane = tid & 31;
  const int wave = tid >> 5;            // 0..7, owns 32 gate columns
  const int bn   = lane & 15;           // B/D column within a 16-wide tile; also A row m
  const int bhi  = lane >> 4;           // K-half / M-half selector
  const int ahi8 = bhi * 8;
  const int colbase = wave * 32;
  const int row0 = blockIdx.x * 16;

  // ---- init LDS h state ----
  for (int i = tid; i < 16 * HSTR; i += 256) { sh0[i] = (_Float16)0; sh1[i] = (_Float16)0; }

  // ---- preload the block's entire x tile to LDS (coalesced, once) ----
  // sxa layout: [t][m][i] = sxa[t*64 + m*4 + i]
  for (int m = 0; m < 16; ++m) {
    const float* xr = x + (size_t)(row0 + m) * (T_STEPS * IN_F);
    for (int idx = tid; idx < T_STEPS * IN_F; idx += 256)
      sxa[(idx >> 2) * 64 + m * 4 + (idx & 3)] = (_Float16)xr[idx];
  }

  // ---- loop-invariant weight B-fragments in registers (f32 -> f16) ----
  // B layout: element (lane, half j): n = lane&15, k = (lane>>4)*16 + j
  V16H B0[2][3];   // layer0: K-chunks [Whh0 k0..31][Whh0 k32..63][Wih0 | zero-pad]
  V16H B1[2][4];   // layer1: [Whh1 x2][Wih1 x2]
  float bias0[2], bias1[2];   // per-lane gate bias for this lane's output column
  #pragma unroll
  for (int tile = 0; tile < 2; ++tile) {
    const int ncol = colbase + tile * 16 + bn;   // gate column 0..255
    bias0[tile] = bih0[ncol] + bhh0[ncol];
    bias1[tile] = bih1[ncol] + bhh1[ncol];
    #pragma unroll
    for (int c = 0; c < 2; ++c)
      #pragma unroll
      for (int j = 0; j < 16; ++j)
        B0[tile][c].e[j] = (_Float16)Whh0[ncol * HID + c * 32 + bhi * 16 + j];
    #pragma unroll
    for (int j = 0; j < 16; ++j) {
      const int kl = bhi * 16 + j;
      B0[tile][2].e[j] = (kl < IN_F) ? (_Float16)Wih0[ncol * IN_F + kl] : (_Float16)0;
    }
    #pragma unroll
    for (int c = 0; c < 2; ++c)
      #pragma unroll
      for (int j = 0; j < 16; ++j)
        B1[tile][c].e[j] = (_Float16)Whh1[ncol * HID + c * 32 + bhi * 16 + j];
    #pragma unroll
    for (int c = 0; c < 2; ++c)
      #pragma unroll
      for (int j = 0; j < 16; ++j)
        B1[tile][2 + c].e[j] = (_Float16)Wih1[ncol * HID + c * 32 + bhi * 16 + j];
  }

  // per-thread cell state: thread owns (m = tid>>4, n = (tid&15)+16q), q=0..3
  float c0[4] = {0.f, 0.f, 0.f, 0.f};
  float c1[4] = {0.f, 0.f, 0.f, 0.f};
  const int em = tid >> 4;
  const int en = tid & 15;

  for (int t = 0; t < T_STEPS; ++t) {
    __syncthreads();   // orders: init/x-preload before 1st use; sG reuse (stage5 -> stage2)

    // ---- stage 2: layer-0 WMMA:  G0 = [h0 | x_t | 0] * [Whh0^T ; Wih0^T ; 0] + bias ----
    {
      const v16h a0 = ld_a(sh0, bn, ahi8, 0);
      const v16h a1 = ld_a(sh0, bn, ahi8, 32);
      V16H a2;
      #pragma unroll
      for (int j = 0; j < 16; ++j) a2.e[j] = (_Float16)0;
      if (bhi == 0) {
        const _Float16* xp = &sxa[t * 64 + bn * 4];
        #pragma unroll
        for (int j = 0; j < IN_F; ++j) a2.e[j] = xp[j];
      }
      #pragma unroll
      for (int tile = 0; tile < 2; ++tile) {
        V8F acc;
        #pragma unroll
        for (int r = 0; r < 8; ++r) acc.e[r] = bias0[tile];   // bias folded into C
        acc.v = wmma16(a0,   B0[tile][0].v, acc.v);
        acc.v = wmma16(a1,   B0[tile][1].v, acc.v);
        acc.v = wmma16(a2.v, B0[tile][2].v, acc.v);
        #pragma unroll
        for (int r = 0; r < 8; ++r)   // D layout: m = r + 8*bhi, n = bn
          sG[(r + 8 * bhi) * GSTR + colbase + tile * 16 + bn] = acc.e[r];
      }
    }
    __syncthreads();

    // ---- stage 3: layer-0 gates/elementwise -> new h0 (f16) ----
    #pragma unroll
    for (int q = 0; q < 4; ++q) {
      const int n = en + q * 16;
      const float gi = sG[em * GSTR +        n];
      const float gf = sG[em * GSTR +  64 +  n];
      const float gg = sG[em * GSTR + 128 +  n];
      const float go = sG[em * GSTR + 192 +  n];
      const float c = fsigmoid(gf) * c0[q] + fsigmoid(gi) * ftanh(gg);
      c0[q] = c;
      sh0[em * HSTR + n] = (_Float16)(fsigmoid(go) * ftanh(c));
    }
    __syncthreads();

    // ---- stage 4: layer-1 WMMA:  G1 = [h1 | y_t] * [Whh1^T ; Wih1^T] + bias ----
    {
      const v16h a0 = ld_a(sh1, bn, ahi8, 0);
      const v16h a1 = ld_a(sh1, bn, ahi8, 32);
      const v16h a2 = ld_a(sh0, bn, ahi8, 0);
      const v16h a3 = ld_a(sh0, bn, ahi8, 32);
      #pragma unroll
      for (int tile = 0; tile < 2; ++tile) {
        V8F acc;
        #pragma unroll
        for (int r = 0; r < 8; ++r) acc.e[r] = bias1[tile];
        acc.v = wmma16(a0, B1[tile][0].v, acc.v);
        acc.v = wmma16(a1, B1[tile][1].v, acc.v);
        acc.v = wmma16(a2, B1[tile][2].v, acc.v);
        acc.v = wmma16(a3, B1[tile][3].v, acc.v);
        #pragma unroll
        for (int r = 0; r < 8; ++r)
          sG[(r + 8 * bhi) * GSTR + colbase + tile * 16 + bn] = acc.e[r];
      }
    }
    __syncthreads();

    // ---- stage 5: layer-1 gates/elementwise -> new h1 (f16) ----
    #pragma unroll
    for (int q = 0; q < 4; ++q) {
      const int n = en + q * 16;
      const float gi = sG[em * GSTR +        n];
      const float gf = sG[em * GSTR +  64 +  n];
      const float gg = sG[em * GSTR + 128 +  n];
      const float go = sG[em * GSTR + 192 +  n];
      const float c = fsigmoid(gf) * c1[q] + fsigmoid(gi) * ftanh(gg);
      c1[q] = c;
      sh1[em * HSTR + n] = (_Float16)(fsigmoid(go) * ftanh(c));
    }
    // next iteration's top barrier orders sh1 writes / sG reuse
  }
  __syncthreads();

  // ---- head: z = relu(h_last @ W1^T + b1), out = z @ W2^T + b2 ----
  float zv[4];
  #pragma unroll
  for (int q = 0; q < 4; ++q) {
    const int n = en + q * 16;
    float s = b1[n];
    #pragma unroll 8
    for (int k = 0; k < HID; ++k)
      s += (float)sh1[em * HSTR + k] * W1[n * HID + k];
    zv[q] = fmaxf(s, 0.f);
  }
  __syncthreads();
  #pragma unroll
  for (int q = 0; q < 4; ++q) sG[em * GSTR + en + q * 16] = zv[q];
  __syncthreads();
  if (tid < 192) {
    const int m = tid / 12, o = tid % 12;
    float s = b2[o];
    #pragma unroll 8
    for (int k = 0; k < HID; ++k)
      s += sG[m * GSTR + k] * W2[o * HID + k];
    out[(size_t)(row0 + m) * 12 + o] = s;
  }
}

extern "C" void kernel_launch(void* const* d_in, const int* in_sizes, int n_in,
                              void* d_out, int out_size, void* d_ws, size_t ws_size,
                              hipStream_t stream) {
  const float* x    = (const float*)d_in[0];
  const float* Wih0 = (const float*)d_in[1];
  const float* Whh0 = (const float*)d_in[2];
  const float* bih0 = (const float*)d_in[3];
  const float* bhh0 = (const float*)d_in[4];
  const float* Wih1 = (const float*)d_in[5];
  const float* Whh1 = (const float*)d_in[6];
  const float* bih1 = (const float*)d_in[7];
  const float* bhh1 = (const float*)d_in[8];
  const float* W1   = (const float*)d_in[9];
  const float* b1   = (const float*)d_in[10];
  const float* W2   = (const float*)d_in[11];
  const float* b2   = (const float*)d_in[12];
  float* out = (float*)d_out;

  const int rows = in_sizes[0] / (T_STEPS * IN_F);   // batch B
  const int grid = rows / 16;                        // 16 sequences per block
  lstm2_fused_kernel<<<grid, 256, 0, stream>>>(
      x, Wih0, Whh0, bih0, bhh0, Wih1, Whh1, bih1, bhh1, W1, b1, W2, b2, out);
}
